// LSTMModel_2456721293896
// MI455X (gfx1250) — matrix operand order repaired and checked by reading, not compiled, not run on hardware
//
#include <hip/hip_runtime.h>

typedef __attribute__((ext_vector_type(16))) _Float16 v16h;
typedef __attribute__((ext_vector_type(8)))  _Float16 v8h;
typedef __attribute__((ext_vector_type(8)))  float    v8f;

#define TSEQ   512
#define HID    50
#define NGATE  200            // 4*H
#define MWG    32             // batch rows per workgroup (2 WMMA M-tiles)
#define NWAVES 13             // N-tiles of 16 covering 208 >= 200
#define NTHREADS (NWAVES * 32)
#define HSTR   136            // hbuf row stride (halves); 272B keeps b128 loads bank-staggered
#define GSTR   212            // gate buffer row stride (floats); avoids half-wave bank collisions
#define XSTR   TSEQ

#define WMMA_F16(A, B, C) \
  __builtin_amdgcn_wmma_f32_16x16x32_f16(false, (A), false, (B), (short)0, (C), false, false)

// Fast sigmoid/tanh: v_exp_f32 + v_rcp_f32 (~1 ulp), no IEEE division sequence.
__device__ __forceinline__ float sigf(float x)   { return __builtin_amdgcn_rcpf(1.0f + __expf(-x)); }
__device__ __forceinline__ float tanh_f(float x) { return 2.0f * sigf(2.0f * x) - 1.0f; }

__global__ __launch_bounds__(NTHREADS) void lstm2_wmma_kernel(
    const float* __restrict__ x,
    const float* __restrict__ W_ih0, const float* __restrict__ W_hh0,
    const float* __restrict__ b_ih0, const float* __restrict__ b_hh0,
    const float* __restrict__ W_ih1, const float* __restrict__ W_hh1,
    const float* __restrict__ b_ih1, const float* __restrict__ b_hh1,
    const float* __restrict__ W_fc,  const float* __restrict__ b_fc,
    float* __restrict__ out)
{
  __shared__ __align__(16) float    xbuf[MWG * XSTR];     // 64 KB    staged input
  __shared__ __align__(16) float    g0buf[MWG * GSTR];    // ~26.5 KB layer-0 gate pre-activations
  __shared__ __align__(16) float    g1buf[MWG * GSTR];    // ~26.5 KB layer-1 gate pre-activations
  __shared__ __align__(32) _Float16 hbuf[MWG * HSTR];     // ~8.5 KB  cols 0..49 = h1, 64..113 = h2, pads 0
  __shared__ __align__(16) float    h2f32[MWG * HID];     // 6.25 KB  final-step h2 (full precision)

  const int tid  = threadIdx.x;
  const int lane = tid & 31;           // wave32
  const int wave = tid >> 5;           // == N-tile index, 0..12
  const int nloc = lane & 15;
  const int hi   = lane >> 4;          // half-wave select
  const int ncol = wave * 16 + nloc;   // global gate column, 0..207 (>=200 -> padding lane)
  const int rowbase = blockIdx.x * MWG;

  // ---- stage this block's x tile into LDS (coalesced) ----
  for (int i = tid; i < MWG * TSEQ; i += NTHREADS)
    xbuf[i] = x[rowbase * TSEQ + i];

  // ---- zero h buffer: h0 = 0 and K-padding columns stay 0 forever ----
  for (int i = tid; i < MWG * HSTR; i += NTHREADS)
    hbuf[i] = (_Float16)0.0f;

  // ---- per-lane scalars for accumulator init ----
  float bias0 = 0.f, w0 = 0.f, bias1 = 0.f;
  if (ncol < NGATE) {
    bias0 = b_ih0[ncol] + b_hh0[ncol];
    w0    = W_ih0[ncol];                       // input dim == 1 -> outer product
    bias1 = b_ih1[ncol] + b_hh1[ncol];
  }

  // ---- resident B fragments (weights), per documented 16-bit B layout:
  //      lane%16 = N column; element e -> K = 8*hi + e (e<8), 16 + 8*hi + (e-8) (e>=8) within each 32-K chunk.
  v16h bf0[2];                                  // layer0: h1 @ W_hh0^T, K padded 50->64
  #pragma unroll
  for (int ch = 0; ch < 2; ++ch) {
    v16h tmp;
    #pragma unroll
    for (int e = 0; e < 16; ++e) {
      int k = ch * 32 + hi * 8 + e + (e >= 8 ? 8 : 0);
      float wv = (ncol < NGATE && k < HID) ? W_hh0[ncol * HID + k] : 0.f;
      tmp[e] = (_Float16)wv;
    }
    bf0[ch] = tmp;
  }
  v16h bf1[4];                                  // layer1 fused: [h1|h2] @ [W_ih1|W_hh1]^T, K=128
  #pragma unroll
  for (int ch = 0; ch < 4; ++ch) {
    v16h tmp;
    #pragma unroll
    for (int e = 0; e < 16; ++e) {
      int k = ch * 32 + hi * 8 + e + (e >= 8 ? 8 : 0);
      float wv = 0.f;
      if (ncol < NGATE) {
        if (k < HID)                      wv = W_ih1[ncol * HID + k];
        else if (k >= 64 && k < 64 + HID) wv = W_hh1[ncol * HID + (k - 64)];
      }
      tmp[e] = (_Float16)wv;
    }
    bf1[ch] = tmp;
  }

  // per-thread f32 cell states (thread owns fixed (row,j) elements all steps)
  float c0s[4] = {0.f, 0.f, 0.f, 0.f};
  float c1s[4] = {0.f, 0.f, 0.f, 0.f};

  union AF { v16h v; v8h h[2]; };

  __syncthreads();

  for (int t = 0; t < TSEQ; ++t) {
    // ===== layer 0 GEMM: gates0 = b0 + x_t * w_ih0 + h1 @ W_hh0^T =====
    {
      v8f c0, c1;                                 // two independent M-tiles -> no WMMA RAW chain
      #pragma unroll
      for (int r = 0; r < 8; ++r) {
        c0[r] = bias0 + xbuf[(8 * hi + r) * XSTR + t] * w0;
        c1[r] = bias0 + xbuf[(16 + 8 * hi + r) * XSTR + t] * w0;
      }
      const _Float16* ar0 = &hbuf[nloc * HSTR + 8 * hi];
      const _Float16* ar1 = &hbuf[(16 + nloc) * HSTR + 8 * hi];
      AF a00, a01, a10, a11;
      a00.h[0] = *(const v8h*)(ar0 +  0); a00.h[1] = *(const v8h*)(ar0 + 16);
      a01.h[0] = *(const v8h*)(ar0 + 32); a01.h[1] = *(const v8h*)(ar0 + 48);
      a10.h[0] = *(const v8h*)(ar1 +  0); a10.h[1] = *(const v8h*)(ar1 + 16);
      a11.h[0] = *(const v8h*)(ar1 + 32); a11.h[1] = *(const v8h*)(ar1 + 48);
      c0 = WMMA_F16(a00.v, bf0[0], c0);
      c1 = WMMA_F16(a10.v, bf0[0], c1);
      c0 = WMMA_F16(a01.v, bf0[1], c0);
      c1 = WMMA_F16(a11.v, bf0[1], c1);
      #pragma unroll
      for (int r = 0; r < 8; ++r) {
        g0buf[(8 * hi + r) * GSTR + ncol]      = c0[r];
        g0buf[(16 + 8 * hi + r) * GSTR + ncol] = c1[r];
      }
    }
    __syncthreads();

    // ===== layer 0 elementwise: c = sig(f)*c + sig(i)*tanh(g); h1 = sig(o)*tanh(c) =====
    #pragma unroll
    for (int it = 0; it < 4; ++it) {
      int idx = tid + it * NTHREADS;
      if (idx < MWG * HID) {
        int row = idx / HID, j = idx - row * HID;
        const float* g = &g0buf[row * GSTR + j];
        float cc = sigf(g[HID]) * c0s[it] + sigf(g[0]) * tanh_f(g[2 * HID]);
        c0s[it] = cc;
        hbuf[row * HSTR + j] = (_Float16)(sigf(g[3 * HID]) * tanh_f(cc));
      }
    }
    __syncthreads();

    // ===== layer 1 fused GEMM: gates1 = b1 + [h1|h2] @ [W_ih1|W_hh1]^T =====
    {
      v8f c0, c1;
      #pragma unroll
      for (int r = 0; r < 8; ++r) { c0[r] = bias1; c1[r] = bias1; }
      const _Float16* ar0 = &hbuf[nloc * HSTR + 8 * hi];
      const _Float16* ar1 = &hbuf[(16 + nloc) * HSTR + 8 * hi];
      #pragma unroll
      for (int ch = 0; ch < 4; ++ch) {
        AF a0, a1;
        a0.h[0] = *(const v8h*)(ar0 + ch * 32);
        a0.h[1] = *(const v8h*)(ar0 + ch * 32 + 16);
        a1.h[0] = *(const v8h*)(ar1 + ch * 32);
        a1.h[1] = *(const v8h*)(ar1 + ch * 32 + 16);
        c0 = WMMA_F16(a0.v, bf1[ch], c0);       // alternate accumulators: hazard-free pairs
        c1 = WMMA_F16(a1.v, bf1[ch], c1);
      }
      #pragma unroll
      for (int r = 0; r < 8; ++r) {
        g1buf[(8 * hi + r) * GSTR + ncol]      = c0[r];
        g1buf[(16 + 8 * hi + r) * GSTR + ncol] = c1[r];
      }
    }
    __syncthreads();

    // ===== layer 1 elementwise (no trailing barrier: next-step edges are covered
    //        by the layer-0 barriers since G0/G1 are separate buffers) =====
    #pragma unroll
    for (int it = 0; it < 4; ++it) {
      int idx = tid + it * NTHREADS;
      if (idx < MWG * HID) {
        int row = idx / HID, j = idx - row * HID;
        const float* g = &g1buf[row * GSTR + j];
        float cc = sigf(g[HID]) * c1s[it] + sigf(g[0]) * tanh_f(g[2 * HID]);
        c1s[it] = cc;
        float hh = sigf(g[3 * HID]) * tanh_f(cc);
        hbuf[row * HSTR + 64 + j] = (_Float16)hh;
        if (t == TSEQ - 1) h2f32[row * HID + j] = hh;   // keep full precision for FC
      }
    }
  }
  __syncthreads();

  // ===== final FC: out = h2_last @ W_fc^T + b_fc =====
  if (tid < MWG) {
    float acc = b_fc[0];
    #pragma unroll 10
    for (int j = 0; j < HID; ++j)
      acc += h2f32[tid * HID + j] * W_fc[j];
    out[rowbase + tid] = acc;
  }
}

extern "C" void kernel_launch(void* const* d_in, const int* in_sizes, int n_in,
                              void* d_out, int out_size, void* d_ws, size_t ws_size,
                              hipStream_t stream) {
  const float* xp   = (const float*)d_in[0];
  const float* Wih0 = (const float*)d_in[1];
  const float* Whh0 = (const float*)d_in[2];
  const float* bih0 = (const float*)d_in[3];
  const float* bhh0 = (const float*)d_in[4];
  const float* Wih1 = (const float*)d_in[5];
  const float* Whh1 = (const float*)d_in[6];
  const float* bih1 = (const float*)d_in[7];
  const float* bhh1 = (const float*)d_in[8];
  const float* Wfc  = (const float*)d_in[9];
  const float* bfc  = (const float*)d_in[10];
  float* out = (float*)d_out;

  dim3 grid(4096 / MWG);       // 128 workgroups of 13 waves
  dim3 block(NTHREADS);        // 416 threads
  hipLaunchKernelGGL(lstm2_wmma_kernel, grid, block, 0, stream,
                     xp, Wih0, Whh0, bih0, bhh0, Wih1, Whh1, bih1, bhh1, Wfc, bfc, out);
}
